// EarthAttention3D_74431783240068
// MI455X (gfx1250) — matrix operand order repaired and checked by reading, not compiled, not run on hardware
//
#include <hip/hip_runtime.h>
#include <hip/hip_bf16.h>
#include <stdint.h>

// ---------- CDNA5 WMMA plumbing (gfx1250, wave32) ----------
typedef __attribute__((ext_vector_type(16))) __bf16 v16bf;
typedef __attribute__((ext_vector_type(8)))  float  v8f;

union FragAB { uint4 q[2]; v16bf v; };

static __device__ __forceinline__ unsigned short f2b(float f) {
  unsigned u = __builtin_bit_cast(unsigned, f);
  u += 0x7FFFu + ((u >> 16) & 1u);           // round-to-nearest-even
  return (unsigned short)(u >> 16);
}

static __device__ __forceinline__ v8f wmma_bf16(const FragAB& a, const FragAB& b, v8f c) {
  // D = A(16x32 bf16) * B(32x16 bf16) + C(16x16 f32)
  return __builtin_amdgcn_wmma_f32_16x16x32_bf16(false, a.v, false, b.v, (short)0, c,
                                                 false, false);
}

// ---------- async global->LDS copy path (ASYNCcnt), with fallback ----------
#if defined(__gfx1250__) && __has_builtin(__builtin_amdgcn_global_load_async_to_lds_b128)
#define HAVE_ASYNC_LDS 1
#else
#define HAVE_ASYNC_LDS 0
#endif

typedef int v4i_raw __attribute__((vector_size(16)));
typedef __attribute__((address_space(1))) v4i_raw* g_v4i_ptr;  // global (AS1)
typedef __attribute__((address_space(3))) v4i_raw* l_v4i_ptr;  // LDS (AS3)

static __device__ __forceinline__ void g2l_b128(void* lds, const void* gsrc) {
#if HAVE_ASYNC_LDS
  __builtin_amdgcn_global_load_async_to_lds_b128((g_v4i_ptr)gsrc, (l_v4i_ptr)lds, 0, 0);
#else
  *(uint4*)lds = *(const uint4*)gsrc;
#endif
}

static __device__ __forceinline__ void wait_async0() {
#if HAVE_ASYNC_LDS
#if __has_builtin(__builtin_amdgcn_s_wait_asynccnt)
  __builtin_amdgcn_s_wait_asynccnt(0);
#else
  asm volatile("s_wait_asynccnt 0x0" ::: "memory");
#endif
#endif
}

// ---------- problem constants ----------
#define HEADS 12
#define LWIN  144            // window length
#define DIM   384
#define HD    32             // head dim
#define BWIN  256            // number of windows (batch)
#define LL    (LWIN * LWIN)  // 20736
#define M_ROWS (BWIN * LWIN) // 36864
#define SCALE_Q 0.17677669529663687f  // 32^-0.5

// ---------- f32 -> bf16 conversion ----------
__global__ void cvt_f32_bf16(const float* __restrict__ in,
                             unsigned short* __restrict__ out, int n) {
  for (int i = blockIdx.x * 256 + threadIdx.x; i < n; i += gridDim.x * 256)
    out[i] = f2b(in[i]);
}

// ---------- dense bias gather: out[(win*12+head)*LL + p] = table[pos[p]*384 + win*12+head] ----------
__global__ void bias_expand(const float* __restrict__ table, const int* __restrict__ pos,
                            float* __restrict__ out, int total) {
  for (int i = blockIdx.x * 256 + threadIdx.x; i < total; i += gridDim.x * 256) {
    int p  = i % LL;
    int wh = i / LL;
    out[i] = table[(size_t)pos[p] * (HEADS * 32) + wh];  // 32 win types * 12 heads = 384
  }
}

// ---------- QKV GEMM: C(36864 x 1152) = Xbf(36864x384) @ Wbf^T, scatter to q/k/v (b,h,L,32) bf16 ----------
__global__ __launch_bounds__(256) void qkv_gemm(
    const unsigned short* __restrict__ xbf,
    const unsigned short* __restrict__ wbf,   // (1152, 384) bf16 row-major
    const float* __restrict__ qkv_b,
    unsigned short* __restrict__ qb,
    unsigned short* __restrict__ kb,
    unsigned short* __restrict__ vb) {
  __shared__ __align__(16) unsigned short lsA[128 * 32];
  __shared__ __align__(16) unsigned short lsB[64 * 32];
  const int tid  = threadIdx.x;
  const int lane = tid & 31, wv = tid >> 5;
  const int half = lane >> 4, l16 = lane & 15;
  const int wm = wv >> 1, wn = wv & 1;              // 4x2 wave grid, 32x32 per wave
  const int row0 = blockIdx.x * 128;
  const int col0 = blockIdx.y * 64;

  v8f acc[2][2];
#pragma unroll
  for (int i = 0; i < 2; i++)
#pragma unroll
    for (int j = 0; j < 2; j++) acc[i][j] = (v8f){0, 0, 0, 0, 0, 0, 0, 0};

  for (int kk = 0; kk < 12; ++kk) {
    const int k0 = kk * 32;
    if (kk) __syncthreads();
    // stage A: 128x32 bf16 = 512 b128 chunks (2 per thread), async global->LDS
#pragma unroll
    for (int rep = 0; rep < 2; ++rep) {
      int i = tid + rep * 256;
      g2l_b128((char*)lsA + 16 * i,
               xbf + (size_t)(row0 + (i >> 2)) * DIM + k0 + 8 * (i & 3));
    }
    // stage B: 64x32 bf16 = 256 b128 chunks
    g2l_b128((char*)lsB + 16 * tid,
             wbf + (size_t)(col0 + (tid >> 2)) * DIM + k0 + 8 * (tid & 3));
    if (kk + 1 < 12) {  // hint next k-tile toward L2 (global_prefetch_b8)
      __builtin_prefetch(xbf + (size_t)(row0 + (tid >> 2)) * DIM + k0 + 32, 0, 1);
      __builtin_prefetch(wbf + (size_t)(col0 + (tid >> 2)) * DIM + k0 + 32, 0, 1);
    }
    wait_async0();
    __syncthreads();

    FragAB a[2], b[2];
#pragma unroll
    for (int im = 0; im < 2; ++im) {
      const unsigned short* ap = lsA + (wm * 32 + im * 16 + l16) * 32;
      a[im].q[0] = *(const uint4*)(ap + 8 * half);
      a[im].q[1] = *(const uint4*)(ap + 16 + 8 * half);
    }
#pragma unroll
    for (int jn = 0; jn < 2; ++jn) {
      const unsigned short* bp = lsB + (wn * 32 + jn * 16 + l16) * 32;
      b[jn].q[0] = *(const uint4*)(bp + 8 * half);
      b[jn].q[1] = *(const uint4*)(bp + 16 + 8 * half);
    }
#pragma unroll
    for (int im = 0; im < 2; ++im)
#pragma unroll
      for (int jn = 0; jn < 2; ++jn) acc[im][jn] = wmma_bf16(a[im], b[jn], acc[im][jn]);
  }

  // epilogue: +bias, scale q, scatter bf16 into (b, h, L, 32)
#pragma unroll
  for (int im = 0; im < 2; ++im) {
#pragma unroll
    for (int jn = 0; jn < 2; ++jn) {
      const int cg = col0 + wn * 32 + jn * 16 + l16;   // 0..1151
      const int which = cg / DIM;                      // 0=q 1=k 2=v
      const int cl = cg - which * DIM;
      const int hh = cl >> 5, dd = cl & 31;
      const float bv = qkv_b[cg];
      const float sc = (which == 0) ? SCALE_Q : 1.0f;
      unsigned short* dst = (which == 0) ? qb : (which == 1) ? kb : vb;
#pragma unroll
      for (int v = 0; v < 8; ++v) {
        const int rg = row0 + wm * 32 + im * 16 + v + 8 * half;
        const int bi = rg / LWIN, li = rg - bi * LWIN;
        dst[(((size_t)(bi * HEADS + hh) * LWIN + li) << 5) + dd] = f2b((acc[im][jn][v] + bv) * sc);
      }
    }
  }
}

// ---------- fused attention: one block per (window, head); 9 waves handle 16 rows each ----------
__global__ __launch_bounds__(288) void attn_kernel(
    const unsigned short* __restrict__ qg,
    const unsigned short* __restrict__ kg,
    const unsigned short* __restrict__ vg,
    const float* __restrict__ biasF,   // (32*12, L, L)
    const float* __restrict__ maskF,   // (256, L, L)
    unsigned short* __restrict__ ctx)  // (b, L, 384) bf16
{
  __shared__ __align__(16) unsigned short lsK[LWIN * HD];   //  9 KB : K rows (m, d)
  __shared__ __align__(16) unsigned short lsVt[HD * 160];   // 10 KB : V^T (d, m) padded m->160
  __shared__ __align__(16) unsigned short lsP[LWIN * 160];  // 45 KB : probs (l, m) padded
  const int tid  = threadIdx.x;
  const int lane = tid & 31, w = tid >> 5;
  const int half = lane >> 4, l16 = lane & 15;
  const int bi = blockIdx.x / HEADS;
  const int h  = blockIdx.x - bi * HEADS;
  const size_t kvbase = (size_t)(bi * HEADS + h) * (LWIN * HD);

  // K tile: contiguous, 576 b128 chunks via the async path
  for (int i = tid; i < (LWIN * HD) / 8; i += 288)
    g2l_b128((char*)lsK + 16 * i, (const char*)(kg + kvbase) + 16 * i);
  // V^T tile: transpose scatter, uint (2 halves) granularity
  for (int i = tid; i < (LWIN * HD) / 2; i += 288) {
    int m = i >> 4, d2 = (i & 15) << 1;
    unsigned pr = *(const unsigned*)(vg + kvbase + (size_t)m * HD + d2);
    lsVt[d2 * 160 + m]       = (unsigned short)(pr & 0xFFFFu);
    lsVt[(d2 + 1) * 160 + m] = (unsigned short)(pr >> 16);
  }
  for (int i = tid; i < HD * 16; i += 288) lsVt[(i >> 4) * 160 + LWIN + (i & 15)] = 0;
  for (int i = tid; i < LWIN * 16; i += 288) lsP[(i >> 4) * 160 + LWIN + (i & 15)] = 0;
  wait_async0();
  __syncthreads();

  // Q A-fragment for this wave's 16 rows (direct from global, ISA 16-bit A layout)
  FragAB a;
  {
    const unsigned short* qp = qg + kvbase + (size_t)(16 * w + l16) * HD;
    a.q[0] = *(const uint4*)(qp + 8 * half);
    a.q[1] = *(const uint4*)(qp + 16 + 8 * half);
  }

  // scores: 9 x (16x16) tiles held in registers (single k-step: K=32)
  v8f s[9];
#pragma unroll
  for (int t = 0; t < 9; ++t) {
    FragAB b;
    const unsigned short* kp = lsK + (t * 16 + l16) * HD;
    b.q[0] = *(const uint4*)(kp + 8 * half);
    b.q[1] = *(const uint4*)(kp + 16 + 8 * half);
    v8f z = (v8f){0, 0, 0, 0, 0, 0, 0, 0};
    s[t] = wmma_bf16(a, b, z);
  }

  // + relative-position bias (window type = b % 32) and shifted-window mask
  const size_t bbase = (size_t)((bi & 31) * HEADS + h) * LL;
  const size_t mbase = (size_t)bi * LL;
#pragma unroll
  for (int t = 0; t < 9; ++t) {
    const int col = t * 16 + l16;
#pragma unroll
    for (int v = 0; v < 8; ++v) {
      const int l = 16 * w + v + 8 * half;
      const size_t o = (size_t)l * LWIN + col;
      s[t][v] += biasF[bbase + o] + maskF[mbase + o];
    }
  }

  // softmax over 144 cols; lanes 0-15 reduce row v, lanes 16-31 reduce row v+8
  float rinv[8];
#pragma unroll
  for (int v = 0; v < 8; ++v) {
    float lm = -3.4e38f;
#pragma unroll
    for (int t = 0; t < 9; ++t) lm = fmaxf(lm, s[t][v]);
#pragma unroll
    for (int m = 1; m < 16; m <<= 1) lm = fmaxf(lm, __shfl_xor(lm, m, 32));
    float ls = 0.f;
#pragma unroll
    for (int t = 0; t < 9; ++t) {
      float e = __expf(s[t][v] - lm);
      s[t][v] = e;
      ls += e;
    }
#pragma unroll
    for (int m = 1; m < 16; m <<= 1) ls += __shfl_xor(ls, m, 32);
    rinv[v] = 1.0f / ls;
  }

  // probs -> LDS (bf16, rows padded to K=160 for the PV GEMM)
#pragma unroll
  for (int t = 0; t < 9; ++t) {
    const int col = t * 16 + l16;
#pragma unroll
    for (int v = 0; v < 8; ++v) {
      const int l = 16 * w + v + 8 * half;
      lsP[l * 160 + col] = f2b(s[t][v] * rinv[v]);
    }
  }
  __syncthreads();

  // out(16x32) = P(16x160) @ V(160x32): 5 k-steps x 2 N-tiles
  v8f acc0 = (v8f){0, 0, 0, 0, 0, 0, 0, 0};
  v8f acc1 = (v8f){0, 0, 0, 0, 0, 0, 0, 0};
#pragma unroll
  for (int ks = 0; ks < 5; ++ks) {
    const int k0 = ks * 32;
    FragAB pa, b0, b1;
    const unsigned short* pp = lsP + (16 * w + l16) * 160 + k0;
    pa.q[0] = *(const uint4*)(pp + 8 * half);
    pa.q[1] = *(const uint4*)(pp + 16 + 8 * half);
    const unsigned short* v0 = lsVt + l16 * 160 + k0;
    b0.q[0] = *(const uint4*)(v0 + 8 * half);
    b0.q[1] = *(const uint4*)(v0 + 16 + 8 * half);
    const unsigned short* v1 = lsVt + (16 + l16) * 160 + k0;
    b1.q[0] = *(const uint4*)(v1 + 8 * half);
    b1.q[1] = *(const uint4*)(v1 + 16 + 8 * half);
    acc0 = wmma_bf16(pa, b0, acc0);
    acc1 = wmma_bf16(pa, b1, acc1);
  }

  // context out: (b, l, h*32 + d) bf16
#pragma unroll
  for (int v = 0; v < 8; ++v) {
    const int l = 16 * w + v + 8 * half;
    const size_t o = ((size_t)bi * LWIN + l) * DIM + h * HD;
    ctx[o + l16]      = f2b(acc0[v]);
    ctx[o + 16 + l16] = f2b(acc1[v]);
  }
}

// ---------- proj GEMM: out(36864x384) f32 = ctx_bf @ proj_w^T + proj_b ----------
__global__ __launch_bounds__(256) void proj_gemm(
    const unsigned short* __restrict__ xbf,
    const unsigned short* __restrict__ wbf,  // (384,384) bf16 row-major
    const float* __restrict__ pbias,
    float* __restrict__ out) {
  __shared__ __align__(16) unsigned short lsA[128 * 32];
  __shared__ __align__(16) unsigned short lsB[64 * 32];
  const int tid  = threadIdx.x;
  const int lane = tid & 31, wv = tid >> 5;
  const int half = lane >> 4, l16 = lane & 15;
  const int wm = wv >> 1, wn = wv & 1;
  const int row0 = blockIdx.x * 128;
  const int col0 = blockIdx.y * 64;

  v8f acc[2][2];
#pragma unroll
  for (int i = 0; i < 2; i++)
#pragma unroll
    for (int j = 0; j < 2; j++) acc[i][j] = (v8f){0, 0, 0, 0, 0, 0, 0, 0};

  for (int kk = 0; kk < 12; ++kk) {
    const int k0 = kk * 32;
    if (kk) __syncthreads();
#pragma unroll
    for (int rep = 0; rep < 2; ++rep) {
      int i = tid + rep * 256;
      g2l_b128((char*)lsA + 16 * i,
               xbf + (size_t)(row0 + (i >> 2)) * DIM + k0 + 8 * (i & 3));
    }
    g2l_b128((char*)lsB + 16 * tid,
             wbf + (size_t)(col0 + (tid >> 2)) * DIM + k0 + 8 * (tid & 3));
    if (kk + 1 < 12) {
      __builtin_prefetch(xbf + (size_t)(row0 + (tid >> 2)) * DIM + k0 + 32, 0, 1);
      __builtin_prefetch(wbf + (size_t)(col0 + (tid >> 2)) * DIM + k0 + 32, 0, 1);
    }
    wait_async0();
    __syncthreads();

    FragAB a[2], b[2];
#pragma unroll
    for (int im = 0; im < 2; ++im) {
      const unsigned short* ap = lsA + (wm * 32 + im * 16 + l16) * 32;
      a[im].q[0] = *(const uint4*)(ap + 8 * half);
      a[im].q[1] = *(const uint4*)(ap + 16 + 8 * half);
    }
#pragma unroll
    for (int jn = 0; jn < 2; ++jn) {
      const unsigned short* bp = lsB + (wn * 32 + jn * 16 + l16) * 32;
      b[jn].q[0] = *(const uint4*)(bp + 8 * half);
      b[jn].q[1] = *(const uint4*)(bp + 16 + 8 * half);
    }
#pragma unroll
    for (int im = 0; im < 2; ++im)
#pragma unroll
      for (int jn = 0; jn < 2; ++jn) acc[im][jn] = wmma_bf16(a[im], b[jn], acc[im][jn]);
  }

#pragma unroll
  for (int im = 0; im < 2; ++im) {
#pragma unroll
    for (int jn = 0; jn < 2; ++jn) {
      const int cg = col0 + wn * 32 + jn * 16 + l16;
      const float bv = pbias[cg];
#pragma unroll
      for (int v = 0; v < 8; ++v) {
        const int rg = row0 + wm * 32 + im * 16 + v + 8 * half;
        out[(size_t)rg * DIM + cg] = acc[im][jn][v] + bv;
      }
    }
  }
}

// ---------- launch ----------
extern "C" void kernel_launch(void* const* d_in, const int* in_sizes, int n_in,
                              void* d_out, int out_size, void* d_ws, size_t ws_size,
                              hipStream_t stream) {
  (void)in_sizes; (void)n_in; (void)out_size; (void)ws_size;
  const float* x          = (const float*)d_in[0];
  const float* mask       = (const float*)d_in[1];
  const float* qkv_w      = (const float*)d_in[2];
  const float* qkv_b      = (const float*)d_in[3];
  const float* proj_w     = (const float*)d_in[4];
  const float* proj_b     = (const float*)d_in[5];
  const float* bias_table = (const float*)d_in[6];
  const int*   pos_idx    = (const int*)d_in[7];
  float* out = (float*)d_out;

  char* ws = (char*)d_ws;
  size_t off = 0;
  auto take = [&](size_t bytes) -> char* {
    char* p = ws + off;
    off += (bytes + 255) & ~(size_t)255;
    return p;
  };
  const size_t NEL = (size_t)M_ROWS * DIM;  // 14,155,776
  unsigned short* xbf  = (unsigned short*)take(NEL * 2);
  unsigned short* qwbf = (unsigned short*)take((size_t)3 * DIM * DIM * 2);
  unsigned short* pwbf = (unsigned short*)take((size_t)DIM * DIM * 2);
  unsigned short* qb   = (unsigned short*)take(NEL * 2);
  unsigned short* kb   = (unsigned short*)take(NEL * 2);
  unsigned short* vb   = (unsigned short*)take(NEL * 2);
  unsigned short* ctx  = (unsigned short*)take(NEL * 2);
  float* biasF         = (float*)take((size_t)32 * HEADS * LL * 4);

  cvt_f32_bf16<<<4096, 256, 0, stream>>>(x, xbf, (int)NEL);
  cvt_f32_bf16<<<512, 256, 0, stream>>>(qkv_w, qwbf, 3 * DIM * DIM);
  cvt_f32_bf16<<<256, 256, 0, stream>>>(proj_w, pwbf, DIM * DIM);
  bias_expand<<<4096, 256, 0, stream>>>(bias_table, pos_idx, biasF, 32 * HEADS * LL);

  qkv_gemm<<<dim3(M_ROWS / 128, (3 * DIM) / 64), 256, 0, stream>>>(xbf, qwbf, qkv_b, qb, kb, vb);
  attn_kernel<<<BWIN * HEADS, 288, 0, stream>>>(qb, kb, vb, biasF, mask, ctx);
  proj_gemm<<<dim3(M_ROWS / 128, DIM / 64), 256, 0, stream>>>(ctx, pwbf, proj_b, out);
}
